// MultiHeadAttentionLayer_56581899158013
// MI455X (gfx1250) — matrix operand order repaired and checked by reading, not compiled, not run on hardware
//
#include <hip/hip_runtime.h>
#include <hip/hip_bf16.h>

// MultiHeadAttentionLayer (graph transformer w/ edge features) for MI455X (gfx1250).
// B=8, N=128, IN_DIM=128, IN_DIM_E=64, H=8, K=64.
// adj is a jnp bool array -> treated as 1 byte per element.

typedef __attribute__((ext_vector_type(16))) _Float16 v16h;
typedef __attribute__((ext_vector_type(8)))  float    v8f;
typedef __attribute__((ext_vector_type(2)))  float    v2f;

#define B_    8
#define N_    128
#define IND_  128
#define INDE_ 64
#define H_    8
#define K_    64
#define HK_   512

// ---------------------------------------------------------------------------
// Kernel 1: convert We/We2 (64 x 512 f32) to f16, stored pre-swizzled in WMMA
// B-fragment order. Fragment (ct, kt): B-tile rows kt*32..kt*32+31 (channels),
// cols ct*16..ct*16+15. B layout used here (contiguous k): lane L holds col n=L%16,
// K-half kh=(L/16)*16; VGPR v holds K = kh+2v, kh+2v+1.
// Flat layout: [((ct*2 + kt)*32 + lane)*16 + elem], elem = 2v+o.
// ---------------------------------------------------------------------------
__global__ void cvt_w_kernel(const float* __restrict__ We,
                             const float* __restrict__ We2,
                             _Float16* __restrict__ W16,
                             _Float16* __restrict__ W216) {
  int t = blockIdx.x * 256 + threadIdx.x;          // 0 .. 32767
  if (t >= 32 * 2 * 32 * 16) return;
  int elem = t & 15;
  int lane = (t >> 4) & 31;
  int kt   = (t >> 9) & 1;
  int ct   = t >> 10;
  int n  = lane & 15;
  int kh = (lane >> 4) * 16;
  int v  = elem >> 1, o = elem & 1;
  int row = kt * 32 + kh + 2 * v + o;              // channel (K-dim)
  int col = ct * 16 + n;                           // output column
  W16[t]  = (_Float16)We [row * HK_ + col];
  W216[t] = (_Float16)We2[row * HK_ + col];
}

// ---------------------------------------------------------------------------
// Kernel 2: projections P[w] = h @ W[w], full f32 via v_wmma_f32_16x16x4_f32.
// w: 0=Qh, 1=Kh*K^-0.5, 2=Vh, 3=Q2, 4=K2*K^-0.5.  h is (B*N=1024, 128).
// Grid (64 m-tiles, 8 n-groups, 5), block 128 (4 waves, 1 n-tile each).
// ---------------------------------------------------------------------------
__global__ void proj_kernel(const float* __restrict__ h,
                            const float* __restrict__ Wq,  const float* __restrict__ Wk,
                            const float* __restrict__ Wv,  const float* __restrict__ Wq2,
                            const float* __restrict__ Wk2, float* __restrict__ P) {
  __shared__ float At[16][IND_];
  int widx = blockIdx.z;
  const float* W;
  float scale = 1.0f;
  switch (widx) {
    case 0: W = Wq;  break;
    case 1: W = Wk;  scale = 0.125f; break;   // K^-0.5 = 1/8
    case 2: W = Wv;  break;
    case 3: W = Wq2; break;
    default: W = Wk2; scale = 0.125f; break;
  }
  float* Po = P + (size_t)widx * (B_ * N_ * HK_);
  int row0 = blockIdx.x * 16;
  for (int f = threadIdx.x; f < 16 * IND_; f += blockDim.x)
    At[f >> 7][f & 127] = h[(size_t)(row0 + (f >> 7)) * IND_ + (f & 127)];
  __syncthreads();

  int wave = threadIdx.x >> 5, lane = threadIdx.x & 31;
  int col0 = (blockIdx.y * 4 + wave) * 16;
  int nn = lane & 15, m = lane & 15;
  int koff = (lane >> 4) * 2;                      // 32-bit A/B layout K split
  v8f c = {};
  for (int kk = 0; kk < IND_; kk += 4) {
    v2f a, b;
    a[0] = At[m][kk + koff];
    a[1] = At[m][kk + koff + 1];
    b[0] = W[(size_t)(kk + koff)     * HK_ + col0 + nn];
    b[1] = W[(size_t)(kk + koff + 1) * HK_ + col0 + nn];
    c = __builtin_amdgcn_wmma_f32_16x16x4_f32(false, a, false, b, (short)0, c, false, false);
  }
  int half = lane >> 4;
#pragma unroll
  for (int r = 0; r < 8; ++r)
    Po[(size_t)(row0 + r + 8 * half) * HK_ + col0 + nn] = c[r] * scale;
}

// ---------------------------------------------------------------------------
// Kernel 3 (main): scores[b,h,i,j]. Block = (j-tile of 16, i, b), 8 waves = 8
// heads. E-tile GEMM in f16 WMMA (16x16x32, f32 acc), both branches, then the
// adj-selected trilinear reduction Σ_k c * Q[i,h,k] * K[j,h,k].
// C-layout: VGPR r -> row j (=r + 8*(lane/16)), lane%16 -> col k-in-tile.
// ---------------------------------------------------------------------------
__global__ void scores_kernel(const float* __restrict__ e,
                              const unsigned char* __restrict__ adj,
                              const _Float16* __restrict__ W16,
                              const _Float16* __restrict__ W216,
                              const float* __restrict__ PQ,  const float* __restrict__ PK,
                              const float* __restrict__ PQ2, const float* __restrict__ PK2,
                              float* __restrict__ scores) {
  __shared__ _Float16 eT[16 * INDE_];
  __shared__ unsigned char adjF[16];
  int jt = blockIdx.x, i = blockIdx.y, b = blockIdx.z;
  int j0 = jt * 16;

  const float* esrc = e + (((size_t)(b * N_ + i)) * N_ + j0) * INDE_;
  for (int f = threadIdx.x; f < 16 * INDE_; f += 256)
    eT[f] = (_Float16)esrc[f];
  if (threadIdx.x < 16)
    adjF[threadIdx.x] = adj[(size_t)(b * N_ + i) * N_ + j0 + threadIdx.x];
  __syncthreads();

  int hh = threadIdx.x >> 5, lane = threadIdx.x & 31;
  int mA = lane & 15, half = lane >> 4;

  // A fragments (16x32 f16): lane half gets K in {0-15} or {16-31}.
  v16h a0, a1;
#pragma unroll
  for (int v = 0; v < 8; ++v) {
    int kl = 16 * half + 2 * v;
    a0[2 * v]     = eT[mA * INDE_ + kl];
    a0[2 * v + 1] = eT[mA * INDE_ + kl + 1];
    a1[2 * v]     = eT[mA * INDE_ + 32 + kl];
    a1[2 * v + 1] = eT[mA * INDE_ + 32 + kl + 1];
  }

  v8f cA[2][4];
  const v16h* Wf0 = (const v16h*)W16;
  const v16h* Wf1 = (const v16h*)W216;
#pragma unroll
  for (int br = 0; br < 2; ++br) {
    const v16h* Wf = br ? Wf1 : Wf0;
#pragma unroll
    for (int t = 0; t < 4; ++t) {
      int ct = hh * 4 + t;                         // column tile in (64x512) W
      v16h b0 = Wf[(size_t)(ct * 2 + 0) * 32 + lane];
      v16h b1 = Wf[(size_t)(ct * 2 + 1) * 32 + lane];
      v8f c = {};
      c = __builtin_amdgcn_wmma_f32_16x16x32_f16(false, a0, false, b0, (short)0, c, false, false);
      c = __builtin_amdgcn_wmma_f32_16x16x32_f16(false, a1, false, b1, (short)0, c, false, false);
      cA[br][t] = c;
    }
  }

  // Epilogue: score[j] = Σ_k E[j,k] * Qsel[i,h,k] * Ksel[j,h,k], branch by adj.
  const float* Qp0 = PQ  + ((size_t)(b * N_ + i)) * HK_ + hh * K_;
  const float* Qp1 = PQ2 + ((size_t)(b * N_ + i)) * HK_ + hh * K_;
  float p[8] = {0, 0, 0, 0, 0, 0, 0, 0};
#pragma unroll
  for (int t = 0; t < 4; ++t) {
    int k = t * 16 + mA;                           // this lane's k within head
    float q0 = Qp0[k];
    float q1 = Qp1[k];
#pragma unroll
    for (int r = 0; r < 8; ++r) {
      int jj = r + 8 * half;
      bool ad = adjF[jj] != 0;                     // ef = adj selects branch
      const float* Kp = ad ? PK : PK2;
      float kv = Kp[((size_t)(b * N_ + j0 + jj)) * HK_ + hh * K_ + k];
      float qv = ad ? q0 : q1;
      float cv = ad ? cA[0][t][r] : cA[1][t][r];
      p[r] += cv * qv * kv;
    }
  }
  // Reduce over the 16 lanes of each half (masks < 16 stay inside the half).
#pragma unroll
  for (int r = 0; r < 8; ++r) {
    p[r] += __shfl_xor(p[r], 1, 32);
    p[r] += __shfl_xor(p[r], 2, 32);
    p[r] += __shfl_xor(p[r], 4, 32);
    p[r] += __shfl_xor(p[r], 8, 32);
  }
  if (mA == 0) {
    float* so = scores + (((size_t)(b * H_ + hh) * N_ + i)) * N_ + j0 + 8 * half;
#pragma unroll
    for (int r = 0; r < 8; ++r) so[r] = p[r];
  }
}

// ---------------------------------------------------------------------------
// Kernel 4: per-(b,h) max over the 128x128 score tile (deterministic).
// ---------------------------------------------------------------------------
__global__ void max_kernel(const float* __restrict__ scores, float* __restrict__ maxbuf) {
  __shared__ float red[256];
  int bh = blockIdx.x;                              // 0..63
  const float* s = scores + (size_t)bh * N_ * N_;
  float m = -INFINITY;
  for (int f = threadIdx.x; f < N_ * N_; f += 256) m = fmaxf(m, s[f]);
  red[threadIdx.x] = m;
  __syncthreads();
  for (int st = 128; st > 0; st >>= 1) {
    if (threadIdx.x < st) red[threadIdx.x] = fmaxf(red[threadIdx.x], red[threadIdx.x + st]);
    __syncthreads();
  }
  if (threadIdx.x == 0) maxbuf[bh] = red[0];
}

// ---------------------------------------------------------------------------
// Kernel 5: w = exp(s - max) * mask_i * mask_j * k_RW; out = (w @ Vh) / denom.
// Block = (i-tile of 16, h, b), 128 threads = 4 waves (one 16-col tile each),
// f32 WMMA 16x16x4 over the j=128 contraction.
// ---------------------------------------------------------------------------
__global__ void out_kernel(const float* __restrict__ scores,
                           const float* __restrict__ maxbuf,
                           const float* __restrict__ kRW,
                           const float* __restrict__ mask,
                           const float* __restrict__ PV,
                           float* __restrict__ out) {
  __shared__ float wT[16][N_];
  __shared__ float rsum[16];
  int it = blockIdx.x, hh = blockIdx.y, b = blockIdx.z;
  int i0 = it * 16;
  float smax = maxbuf[b * H_ + hh];
  if (threadIdx.x < 16) rsum[threadIdx.x] = 0.0f;
  __syncthreads();

  const float* srow = scores + (((size_t)(b * H_ + hh)) * N_ + i0) * N_;
  int rr = threadIdx.x >> 3;                       // 0..15 (row of tile)
  int c0 = (threadIdx.x & 7) * 16;                 // 16 cols per thread
  float mi = mask[b * N_ + i0 + rr];
  float lsum = 0.0f;
  for (int u = 0; u < 16; ++u) {
    int j = c0 + u;
    float w = expf(srow[(size_t)rr * N_ + j] - smax);
    w *= mi * mask[b * N_ + j];
    w *= kRW[((size_t)b * N_ + (i0 + rr)) * N_ + j];
    wT[rr][j] = w;
    lsum += w;
  }
  atomicAdd(&rsum[rr], lsum);
  __syncthreads();

  int wave = threadIdx.x >> 5, lane = threadIdx.x & 31;
  int col0 = wave * 16;                            // within K_=64
  int m = lane & 15, nn = lane & 15, half = lane >> 4;
  int koff = half * 2;
  v8f c = {};
  for (int kk = 0; kk < N_; kk += 4) {
    v2f a, bb;
    a[0] = wT[m][kk + koff];
    a[1] = wT[m][kk + koff + 1];
    bb[0] = PV[((size_t)(b * N_ + kk + koff))     * HK_ + hh * K_ + col0 + nn];
    bb[1] = PV[((size_t)(b * N_ + kk + koff + 1)) * HK_ + hh * K_ + col0 + nn];
    c = __builtin_amdgcn_wmma_f32_16x16x4_f32(false, a, false, bb, (short)0, c, false, false);
  }
#pragma unroll
  for (int r = 0; r < 8; ++r) {
    int row = r + 8 * half;
    float d = fmaxf(rsum[row], 1e-6f);             // clip(denom, 1e-6)
    out[((size_t)(b * N_ + i0 + row)) * HK_ + hh * K_ + col0 + nn] = c[r] / d;
  }
}

// ---------------------------------------------------------------------------
extern "C" void kernel_launch(void* const* d_in, const int* in_sizes, int n_in,
                              void* d_out, int out_size, void* d_ws, size_t ws_size,
                              hipStream_t stream) {
  const float* h    = (const float*)d_in[0];
  const float* e    = (const float*)d_in[1];
  const float* kRW  = (const float*)d_in[2];
  const float* mask = (const float*)d_in[3];
  const unsigned char* adj = (const unsigned char*)d_in[4];   // jnp bool
  const float* Wq   = (const float*)d_in[5];
  const float* Wk   = (const float*)d_in[6];
  const float* Wv   = (const float*)d_in[7];
  const float* We   = (const float*)d_in[8];
  const float* Wq2  = (const float*)d_in[9];
  const float* Wk2  = (const float*)d_in[10];
  const float* We2  = (const float*)d_in[11];
  float* out = (float*)d_out;

  // Workspace layout (all offsets 256B aligned). Total ~14.9 MB.
  char* ws = (char*)d_ws;
  size_t off = 0;
  float* P = (float*)(ws + off);                 // 5 x (1024 x 512) f32
  off += 5ull * B_ * N_ * HK_ * sizeof(float);
  float* scores = (float*)(ws + off);            // (B,H,N,N) f32 = 4 MB
  off += (size_t)B_ * H_ * N_ * N_ * sizeof(float);
  float* maxbuf = (float*)(ws + off);
  off += 256;
  _Float16* W16 = (_Float16*)(ws + off);         // 64 KB swizzled f16 We
  off += 32768 * sizeof(_Float16);
  _Float16* W216 = (_Float16*)(ws + off);        // 64 KB swizzled f16 We2
  off += 32768 * sizeof(_Float16);

  const float* PQ  = P + 0ull * B_ * N_ * HK_;
  const float* PK  = P + 1ull * B_ * N_ * HK_;
  const float* PV  = P + 2ull * B_ * N_ * HK_;
  const float* PQ2 = P + 3ull * B_ * N_ * HK_;
  const float* PK2 = P + 4ull * B_ * N_ * HK_;

  cvt_w_kernel<<<128, 256, 0, stream>>>(We, We2, W16, W216);
  proj_kernel<<<dim3(64, 8, 5), 128, 0, stream>>>(h, Wq, Wk, Wv, Wq2, Wk2, P);
  scores_kernel<<<dim3(N_ / 16, N_, B_), 256, 0, stream>>>(e, adj, W16, W216,
                                                           PQ, PK, PQ2, PK2, scores);
  max_kernel<<<B_ * H_, 256, 0, stream>>>(scores, maxbuf);
  out_kernel<<<dim3(N_ / 16, H_, B_), 128, 0, stream>>>(scores, maxbuf, kRW, mask, PV, out);
}
